// RWKV7TimeMix_83958020702210
// MI455X (gfx1250) — matrix operand order repaired and checked by reading, not compiled, not run on hardware
//
#include <hip/hip_runtime.h>
#include <hip/hip_bf16.h>
#include <cstdint>

typedef __bf16 bf16_t;
typedef __attribute__((ext_vector_type(16))) __bf16 v16bf;
typedef __attribute__((ext_vector_type(8)))  __bf16 v8bf;
typedef __attribute__((ext_vector_type(8)))  float  v8f;

#define H_    2048
#define NH_   32
#define HS_   64
#define B_    2
#define T_    4096
#define NTOK_ (B_ * T_)               // 8192
static const size_t NE_ = (size_t)NTOK_ * H_;   // 16,777,216

__device__ __forceinline__ float sigf(float x) { return 1.0f / (1.0f + __expf(-x)); }

// ---------------------------------------------------------------------------
// Token-shift mixing: produce six bf16 mixed inputs
// ---------------------------------------------------------------------------
__global__ void mix6_kernel(const float* __restrict__ x, const float* __restrict__ shift,
                            const float* __restrict__ cr, const float* __restrict__ cw,
                            const float* __restrict__ ck, const float* __restrict__ cv,
                            const float* __restrict__ ca, const float* __restrict__ cg,
                            bf16_t* __restrict__ mr, bf16_t* __restrict__ mw,
                            bf16_t* __restrict__ mk, bf16_t* __restrict__ mv,
                            bf16_t* __restrict__ ma, bf16_t* __restrict__ mg) {
  const int tok = blockIdx.x;
  const int b = tok / T_, t = tok % T_;
  const float* xrow = x + (size_t)tok * H_;
  const float* prow = (t == 0) ? (shift + (size_t)b * H_) : (xrow - H_);
  for (int c = threadIdx.x; c < H_; c += blockDim.x) {
    const float xc = xrow[c];
    const float dx = prow[c] - xc;
    const size_t o = (size_t)tok * H_ + c;
    mr[o] = (bf16_t)(xc + dx * cr[c]);
    mw[o] = (bf16_t)(xc + dx * cw[c]);
    mk[o] = (bf16_t)(xc + dx * ck[c]);
    mv[o] = (bf16_t)(xc + dx * cv[c]);
    ma[o] = (bf16_t)(xc + dx * ca[c]);
    mg[o] = (bf16_t)(xc + dx * cg[c]);
  }
}

// ---------------------------------------------------------------------------
// f32 -> bf16 casts (with optional activation)
// ---------------------------------------------------------------------------
__global__ void cast_bf16_kernel(const float* __restrict__ s, bf16_t* __restrict__ d, size_t n) {
  size_t i = (size_t)blockIdx.x * blockDim.x + threadIdx.x;
  if (i < n) d[i] = (bf16_t)s[i];
}
__global__ void tanh_bf16_kernel(const float* __restrict__ s, bf16_t* __restrict__ d, size_t n) {
  size_t i = (size_t)blockIdx.x * blockDim.x + threadIdx.x;
  if (i < n) d[i] = (bf16_t)tanhf(s[i]);
}
__global__ void sig_bf16_kernel(const float* __restrict__ s, bf16_t* __restrict__ d, size_t n) {
  size_t i = (size_t)blockIdx.x * blockDim.x + threadIdx.x;
  if (i < n) d[i] = (bf16_t)sigf(s[i]);
}

// src is [Ks, Ns] row-major f32; dst is [Nd, Kd] row-major bf16, zero padded.
__global__ void transpose_pad_kernel(const float* __restrict__ s, bf16_t* __restrict__ d,
                                     int Ks, int Ns, int Kd, int Nd) {
  int i = blockIdx.x * blockDim.x + threadIdx.x;
  if (i >= Nd * Kd) return;
  int n = i / Kd, k = i % Kd;
  float v = (n < Ns && k < Ks) ? s[(size_t)k * Ns + n] : 0.0f;
  d[i] = (bf16_t)v;
}

// ---------------------------------------------------------------------------
// Tensor Data Mover: 2D tile (32 rows x 32 bf16) global -> LDS, with LDS
// padding of 16B after every 64B so LDS row stride is 80B (LDA_=40 bf16).
// ---------------------------------------------------------------------------
#define LDA_ 40   // bf16 elements per LDS row (80 B: 16B aligned, conflict-free)

#if __has_builtin(__builtin_amdgcn_tensor_load_to_lds)
#define USE_TDM_ 1
typedef __attribute__((ext_vector_type(4))) unsigned int u32x4_t;
typedef __attribute__((ext_vector_type(4))) int i32x4_t;
typedef __attribute__((ext_vector_type(8))) int i32x8_t;

__device__ __forceinline__ void tdm_load_tile2d(unsigned lds_off, const void* gptr, unsigned K) {
  const unsigned long long ga = (unsigned long long)(uintptr_t)gptr;
  u32x4_t g0;
  g0[0] = 1u;                                                 // count=1, user D#
  g0[1] = lds_off;                                            // LDS byte address
  g0[2] = (unsigned)ga;                                       // global_addr[31:0]
  g0[3] = (unsigned)((ga >> 32) & 0x01ffffffu) | (2u << 30);  // addr[56:32] | type=2
  i32x8_t g1;
  // data_size=1 (2B) | pad_enable | pad_interval=3 (16 DW) | pad_amount=3 (4 DW)
  g1[0] = (int)((1u << 16) | (1u << 20) | (3u << 22) | (3u << 25));
  g1[1] = (int)((K & 0xffffu) << 16);                         // tensor_dim0[15:0]
  g1[2] = (int)(((K >> 16) & 0xffffu) | (32u << 16));         // dim0[31:16] | tensor_dim1=32
  g1[3] = (int)(32u << 16);                                   // tensor_dim1 hi | tile_dim0=32
  g1[4] = 32;                                                 // tile_dim1=32, tile_dim2=0
  g1[5] = (int)K;                                             // tensor_dim0_stride[31:0]
  g1[6] = 0;                                                  // stride hi | dim1_stride lo
  g1[7] = 0;
  i32x4_t z4; z4[0] = 0; z4[1] = 0; z4[2] = 0; z4[3] = 0;
  i32x8_t z8;
#pragma unroll
  for (int q = 0; q < 8; ++q) z8[q] = 0;
  __builtin_amdgcn_tensor_load_to_lds(g0, g1, z4, z4, z8, 0);
}
#else
#define USE_TDM_ 0
#endif

// ---------------------------------------------------------------------------
// WMMA bf16 GEMM:  C[M,N] = A[M,K] * W[N,K]^T   (f32 accumulate)
// block: 256 threads (8 waves), tile 128x128, BK=32, TDM double-buffered LDS
// requires M%128==0, N%128==0, K%32==0
// ---------------------------------------------------------------------------
__global__ __launch_bounds__(256) void gemm_bf16_wmma(
    const bf16_t* __restrict__ A, const bf16_t* __restrict__ Bw,
    float* __restrict__ C, int M, int N, int K) {
  __shared__ bf16_t sA[2][128 * LDA_];
  __shared__ bf16_t sB[2][128 * LDA_];
  const int tid  = threadIdx.x;
  const int lane = tid & 31;
  const int wave = tid >> 5;
  const int waveM = (wave & 1) * 64;   // 2 waves along M (64 rows each)
  const int waveN = (wave >> 1) * 32;  // 4 waves along N (32 cols each)
  const int blockM = blockIdx.y * 128;
  const int blockN = blockIdx.x * 128;

  v8f acc[4][2];
#pragma unroll
  for (int mt = 0; mt < 4; ++mt)
#pragma unroll
    for (int nt = 0; nt < 2; ++nt)
#pragma unroll
      for (int i = 0; i < 8; ++i) acc[mt][nt][i] = 0.0f;

  const int lrow  = lane & 15;
  const int abase = (lane < 16) ? 0 : 8;    // A: K chunk base per lane half
  const int khalf = (lane < 16) ? 0 : 8;    // B: K chunk base per lane half
  const int nk = K / 32;

#if USE_TDM_
  // DMA work split: waves 0-3 stage A rows [wave*32,+32), waves 4-7 stage B.
  const bool doA = (wave < 4);
  const int  rs  = (doA ? wave : wave - 4) * 32;
  const bf16_t* gsrc = doA ? (A + (size_t)(blockM + rs) * K)
                           : (Bw + (size_t)(blockN + rs) * K);
  bf16_t* lds0 = doA ? &sA[0][rs * LDA_] : &sB[0][rs * LDA_];
  bf16_t* lds1 = doA ? &sA[1][rs * LDA_] : &sB[1][rs * LDA_];

  tdm_load_tile2d((unsigned)(uintptr_t)lds0, gsrc, (unsigned)K);   // tile 0

  for (int i = 0; i < nk; ++i) {
    const int buf = i & 1;
    __builtin_amdgcn_s_wait_tensorcnt(0);    // this wave's chunk of tile i done
    __syncthreads();                          // all chunks of tile i visible
    if (i + 1 < nk)                           // overlap DMA(i+1) with compute(i)
      tdm_load_tile2d((unsigned)(uintptr_t)(buf ? lds0 : lds1),
                      gsrc + (size_t)(i + 1) * 32, (unsigned)K);

    v16bf af[4], bfr[2];
#pragma unroll
    for (int mt = 0; mt < 4; ++mt) {
      const int r = waveM + mt * 16 + lrow;
      const v8bf lo = *(const v8bf*)(&sA[buf][r * LDA_ + abase]);
      const v8bf hi = *(const v8bf*)(&sA[buf][r * LDA_ + abase + 16]);
#pragma unroll
      for (int q = 0; q < 8; ++q) { af[mt][q] = lo[q]; af[mt][q + 8] = hi[q]; }
    }
#pragma unroll
    for (int nt = 0; nt < 2; ++nt) {
      const int r = waveN + nt * 16 + lrow;
      const v8bf lo = *(const v8bf*)(&sB[buf][r * LDA_ + khalf]);
      const v8bf hi = *(const v8bf*)(&sB[buf][r * LDA_ + khalf + 16]);
#pragma unroll
      for (int q = 0; q < 8; ++q) { bfr[nt][q] = lo[q]; bfr[nt][q + 8] = hi[q]; }
    }
#pragma unroll
    for (int mt = 0; mt < 4; ++mt)
#pragma unroll
      for (int nt = 0; nt < 2; ++nt)
        acc[mt][nt] = __builtin_amdgcn_wmma_f32_16x16x32_bf16(
            false, af[mt], false, bfr[nt], (short)0, acc[mt][nt], false, false);
  }
#else
  for (int i = 0; i < nk; ++i) {
    const int k0 = i * 32;
    for (int ch = tid; ch < 512; ch += 256) {
      const int row = ch >> 2, c = ch & 3;
      *(v8bf*)(&sA[0][row * LDA_ + c * 8]) =
          *(const v8bf*)(A + (size_t)(blockM + row) * K + k0 + c * 8);
      *(v8bf*)(&sB[0][row * LDA_ + c * 8]) =
          *(const v8bf*)(Bw + (size_t)(blockN + row) * K + k0 + c * 8);
    }
    __syncthreads();
    v16bf af[4], bfr[2];
#pragma unroll
    for (int mt = 0; mt < 4; ++mt) {
      const int r = waveM + mt * 16 + lrow;
      const v8bf lo = *(const v8bf*)(&sA[0][r * LDA_ + abase]);
      const v8bf hi = *(const v8bf*)(&sA[0][r * LDA_ + abase + 16]);
#pragma unroll
      for (int q = 0; q < 8; ++q) { af[mt][q] = lo[q]; af[mt][q + 8] = hi[q]; }
    }
#pragma unroll
    for (int nt = 0; nt < 2; ++nt) {
      const int r = waveN + nt * 16 + lrow;
      const v8bf lo = *(const v8bf*)(&sB[0][r * LDA_ + khalf]);
      const v8bf hi = *(const v8bf*)(&sB[0][r * LDA_ + khalf + 16]);
#pragma unroll
      for (int q = 0; q < 8; ++q) { bfr[nt][q] = lo[q]; bfr[nt][q + 8] = hi[q]; }
    }
#pragma unroll
    for (int mt = 0; mt < 4; ++mt)
#pragma unroll
      for (int nt = 0; nt < 2; ++nt)
        acc[mt][nt] = __builtin_amdgcn_wmma_f32_16x16x32_bf16(
            false, af[mt], false, bfr[nt], (short)0, acc[mt][nt], false, false);
    __syncthreads();
  }
#endif

  // store: D f32 layout — VGPR r: M = r + 8*(lane/16); N = lane%16
#pragma unroll
  for (int mt = 0; mt < 4; ++mt)
#pragma unroll
    for (int nt = 0; nt < 2; ++nt)
#pragma unroll
      for (int r = 0; r < 8; ++r) {
        const int row = blockM + waveM + mt * 16 + r + ((lane >> 4) << 3);
        const int col = blockN + waveN + nt * 16 + (lane & 15);
        C[(size_t)row * N + col] = acc[mt][nt][r];
      }
}

// ---------------------------------------------------------------------------
// elementwise finishers
// ---------------------------------------------------------------------------
__global__ void decay_kernel(float* __restrict__ wdec, const float* __restrict__ w0, size_t n) {
  size_t i = (size_t)blockIdx.x * blockDim.x + threadIdx.x;
  if (i >= n) return;
  const int c = (int)(i % H_);
  const float u = w0[c] + wdec[i];
  const float wf = -log1pf(__expf(-u)) - 0.5f;     // -softplus(-u) - 0.5
  wdec[i] = __expf(-__expf(wf));
}

__global__ void afinal_kernel(float* __restrict__ a, const float* __restrict__ a0, size_t n) {
  size_t i = (size_t)blockIdx.x * blockDim.x + threadIdx.x;
  if (i >= n) return;
  a[i] = sigf(a0[(int)(i % H_)] + a[i]);
}

__global__ void vupdate_kernel(float* __restrict__ v, const float* __restrict__ vs,
                               const float* __restrict__ vf, const float* __restrict__ v0, size_t n) {
  size_t i = (size_t)blockIdx.x * blockDim.x + threadIdx.x;
  if (i >= n) return;
  const float g = sigf(v0[(int)(i % H_)] + vs[i]);
  v[i] = v[i] + (vf[i] - v[i]) * g;
}

// per-head: kk = normalize(k*k_k), k = k*(1+(a-1)*k_a)
__global__ __launch_bounds__(64) void kk_kernel(float* __restrict__ k, float* __restrict__ kkout,
                                                const float* __restrict__ a,
                                                const float* __restrict__ k_k,
                                                const float* __restrict__ k_a) {
  const int g  = blockIdx.x;          // token*NH + h
  const int h  = g % NH_;
  const int tx = threadIdx.x;
  const int c  = h * HS_ + tx;
  const size_t o = (size_t)(g / NH_) * H_ + c;
  const float kv  = k[o];
  const float kkv = kv * k_k[c];
  __shared__ float red[HS_];
  red[tx] = kkv * kkv;
  __syncthreads();
  for (int s = 32; s > 0; s >>= 1) {
    if (tx < s) red[tx] += red[tx + s];
    __syncthreads();
  }
  const float norm = fmaxf(sqrtf(red[0]), 1e-12f);
  kkout[o] = kkv / norm;
  k[o] = kv * (1.0f + (a[o] - 1.0f) * k_a[c]);
}

// ---------------------------------------------------------------------------
// sequential WKV scan: one block per (b, head); thread v owns state row S[v][:]
// ---------------------------------------------------------------------------
__global__ __launch_bounds__(64) void wkv_scan_kernel(
    const float* __restrict__ rb, const float* __restrict__ wb,
    const float* __restrict__ kb, const float* __restrict__ vb,
    const float* __restrict__ kkb, const float* __restrict__ ab,
    const float* __restrict__ s_in, float* __restrict__ yb, float* __restrict__ s_out) {
  const int bh = blockIdx.x;            // b*NH + h
  const int b = bh / NH_, h = bh % NH_;
  const int v = threadIdx.x;            // value row index 0..63
  float S[HS_];
  const size_t sbase = ((size_t)bh * HS_ + v) * HS_;
#pragma unroll
  for (int k = 0; k < HS_; ++k) S[k] = s_in[sbase + k];

  __shared__ float lr[HS_], lw[HS_], lk[HS_], lv[HS_], lkk[HS_], la[HS_];

  for (int t = 0; t < T_; ++t) {
    const size_t o = ((size_t)b * T_ + t) * H_ + h * HS_ + v;
    lr[v] = rb[o]; lw[v] = wb[o]; lk[v] = kb[o];
    lv[v] = vb[o]; lkk[v] = kkb[o]; la[v] = ab[o];
    __syncthreads();
    float sa = 0.0f;
#pragma unroll
    for (int k = 0; k < HS_; ++k) sa -= S[k] * lkk[k];
    const float vt = lv[v];
    float y = 0.0f;
#pragma unroll
    for (int k = 0; k < HS_; ++k) {
      S[k] = S[k] * lw[k] + sa * (lkk[k] * la[k]) + vt * lk[k];
      y += S[k] * lr[k];
    }
    yb[o] = y;
    __syncthreads();
  }
#pragma unroll
  for (int k = 0; k < HS_; ++k) s_out[sbase + k] = S[k];
}

// ---------------------------------------------------------------------------
// GroupNorm + rk*v residual + output gate -> bf16 for final GEMM
// ---------------------------------------------------------------------------
__global__ __launch_bounds__(64) void gn_out_kernel(
    const float* __restrict__ yb, const float* __restrict__ rb,
    const float* __restrict__ kb, const float* __restrict__ vb,
    const float* __restrict__ gb, const float* __restrict__ r_k,
    const float* __restrict__ ln_w, const float* __restrict__ ln_b,
    bf16_t* __restrict__ yg) {
  const int g  = blockIdx.x;            // token*NH + h
  const int h  = g % NH_;
  const int tx = threadIdx.x;
  const int c  = h * HS_ + tx;
  const size_t o = (size_t)(g / NH_) * H_ + c;
  const float yv = yb[o];
  __shared__ float red[3][HS_];
  red[0][tx] = yv;
  red[1][tx] = yv * yv;
  red[2][tx] = rb[o] * kb[o] * r_k[h * HS_ + tx];
  __syncthreads();
  for (int s = 32; s > 0; s >>= 1) {
    if (tx < s) {
      red[0][tx] += red[0][tx + s];
      red[1][tx] += red[1][tx + s];
      red[2][tx] += red[2][tx + s];
    }
    __syncthreads();
  }
  const float mu  = red[0][0] * (1.0f / HS_);
  const float var = red[1][0] * (1.0f / HS_) - mu * mu;
  const float rk  = red[2][0];
  float yn = (yv - mu) * rsqrtf(var + 1e-5f * HS_) * ln_w[c] + ln_b[c];
  yn += rk * vb[o];
  yg[o] = (bf16_t)(yn * gb[o]);
}

__global__ void copy_lastx_kernel(const float* __restrict__ x, float* __restrict__ d) {
  int i = blockIdx.x * blockDim.x + threadIdx.x;
  if (i >= B_ * H_) return;
  const int b = i / H_, c = i % H_;
  d[i] = x[((size_t)b * T_ + (T_ - 1)) * H_ + c];
}

__global__ void copy_f32_kernel(const float* __restrict__ s, float* __restrict__ d, size_t n) {
  size_t i = (size_t)blockIdx.x * blockDim.x + threadIdx.x;
  if (i < n) d[i] = s[i];
}

// ---------------------------------------------------------------------------
// host driver
// ---------------------------------------------------------------------------
extern "C" void kernel_launch(void* const* d_in, const int* in_sizes, int n_in,
                              void* d_out, int out_size, void* d_ws, size_t ws_size,
                              hipStream_t stream) {
  (void)in_sizes; (void)n_in; (void)out_size; (void)ws_size;
  const float* x      = (const float*)d_in[0];
  const float* shift  = (const float*)d_in[1];
  const float* wkv_in = (const float*)d_in[2];
  const float* vfirst = (const float*)d_in[3];
  const float* c_r = (const float*)d_in[4];
  const float* c_w = (const float*)d_in[5];
  const float* c_k = (const float*)d_in[6];
  const float* c_v = (const float*)d_in[7];
  const float* c_a = (const float*)d_in[8];
  const float* c_g = (const float*)d_in[9];
  const float* w0 = (const float*)d_in[10];
  const float* w1 = (const float*)d_in[11];
  const float* w2 = (const float*)d_in[12];
  const float* a0 = (const float*)d_in[13];
  const float* a1 = (const float*)d_in[14];
  const float* a2 = (const float*)d_in[15];
  const float* v0 = (const float*)d_in[16];
  const float* v1 = (const float*)d_in[17];
  const float* v2 = (const float*)d_in[18];
  const float* g1 = (const float*)d_in[19];
  const float* g2 = (const float*)d_in[20];
  const float* k_k = (const float*)d_in[21];
  const float* k_a = (const float*)d_in[22];
  const float* r_k = (const float*)d_in[23];
  const float* W_r = (const float*)d_in[24];
  const float* W_k = (const float*)d_in[25];
  const float* W_v = (const float*)d_in[26];
  const float* W_o = (const float*)d_in[27];
  const float* ln_w = (const float*)d_in[28];
  const float* ln_b = (const float*)d_in[29];
  float* dout = (float*)d_out;

  // --- workspace bump allocator (256B aligned) ---
  char* wsp = (char*)d_ws;
  auto alloc = [&](size_t bytes) -> void* {
    void* p = (void*)wsp;
    wsp += (bytes + 255) & ~(size_t)255;
    return p;
  };
  const size_t NEb2 = NE_ * sizeof(bf16_t);
  const size_t NEf4 = NE_ * sizeof(float);
  bf16_t* mr = (bf16_t*)alloc(NEb2);
  bf16_t* mw = (bf16_t*)alloc(NEb2);
  bf16_t* mk = (bf16_t*)alloc(NEb2);
  bf16_t* mv = (bf16_t*)alloc(NEb2);
  bf16_t* ma = (bf16_t*)alloc(NEb2);
  bf16_t* mg = (bf16_t*)alloc(NEb2);
  float* rbuf  = (float*)alloc(NEf4);
  float* kbuf  = (float*)alloc(NEf4);
  float* vbuf  = (float*)alloc(NEf4);
  float* wdec  = (float*)alloc(NEf4);
  float* kkbuf = (float*)alloc(NEf4);
  float* abuf  = (float*)alloc(NEf4);
  float* gbuf  = (float*)alloc(NEf4);
  float* ybuf  = (float*)alloc(NEf4);
  float* tmpE  = (float*)alloc(NEf4);
  float*  t1   = (float*)alloc((size_t)NTOK_ * 256 * sizeof(float));
  bf16_t* t1b  = (bf16_t*)alloc((size_t)NTOK_ * 256 * sizeof(bf16_t));
  bf16_t* yg   = (bf16_t*)alloc(NEb2);
  bf16_t* Wr_b = (bf16_t*)alloc((size_t)H_ * H_ * 2);
  bf16_t* Wk_b = (bf16_t*)alloc((size_t)H_ * H_ * 2);
  bf16_t* Wv_b = (bf16_t*)alloc((size_t)H_ * H_ * 2);
  bf16_t* Wo_b = (bf16_t*)alloc((size_t)H_ * H_ * 2);
  bf16_t* w1p = (bf16_t*)alloc((size_t)128 * H_ * 2);
  bf16_t* w2p = (bf16_t*)alloc((size_t)H_ * 128 * 2);
  bf16_t* a1p = (bf16_t*)alloc((size_t)128 * H_ * 2);
  bf16_t* a2p = (bf16_t*)alloc((size_t)H_ * 128 * 2);
  bf16_t* v1p = (bf16_t*)alloc((size_t)128 * H_ * 2);
  bf16_t* v2p = (bf16_t*)alloc((size_t)H_ * 128 * 2);
  bf16_t* g1p = (bf16_t*)alloc((size_t)256 * H_ * 2);
  bf16_t* g2p = (bf16_t*)alloc((size_t)H_ * 256 * 2);

  const size_t HH = (size_t)H_ * H_;
  const int TB = 256;
  auto g1d = [&](size_t n) { return dim3((unsigned)((n + TB - 1) / TB)); };

  // weight conversions
  cast_bf16_kernel<<<g1d(HH), TB, 0, stream>>>(W_r, Wr_b, HH);
  cast_bf16_kernel<<<g1d(HH), TB, 0, stream>>>(W_k, Wk_b, HH);
  cast_bf16_kernel<<<g1d(HH), TB, 0, stream>>>(W_v, Wv_b, HH);
  cast_bf16_kernel<<<g1d(HH), TB, 0, stream>>>(W_o, Wo_b, HH);
  transpose_pad_kernel<<<g1d((size_t)128 * H_), TB, 0, stream>>>(w1, w1p, H_, 96, H_, 128);
  transpose_pad_kernel<<<g1d((size_t)H_ * 128), TB, 0, stream>>>(w2, w2p, 96, H_, 128, H_);
  transpose_pad_kernel<<<g1d((size_t)128 * H_), TB, 0, stream>>>(a1, a1p, H_, 96, H_, 128);
  transpose_pad_kernel<<<g1d((size_t)H_ * 128), TB, 0, stream>>>(a2, a2p, 96, H_, 128, H_);
  transpose_pad_kernel<<<g1d((size_t)128 * H_), TB, 0, stream>>>(v1, v1p, H_, 64, H_, 128);
  transpose_pad_kernel<<<g1d((size_t)H_ * 128), TB, 0, stream>>>(v2, v2p, 64, H_, 128, H_);
  transpose_pad_kernel<<<g1d((size_t)256 * H_), TB, 0, stream>>>(g1, g1p, H_, 256, H_, 256);
  transpose_pad_kernel<<<g1d((size_t)H_ * 256), TB, 0, stream>>>(g2, g2p, 256, H_, 256, H_);

  // token-shift mixing
  mix6_kernel<<<NTOK_, TB, 0, stream>>>(x, shift, c_r, c_w, c_k, c_v, c_a, c_g,
                                        mr, mw, mk, mv, ma, mg);

  // big projections (WMMA): r/k/v
  dim3 gBig(H_ / 128, NTOK_ / 128);
  gemm_bf16_wmma<<<gBig, 256, 0, stream>>>(mr, Wr_b, rbuf, NTOK_, H_, H_);
  gemm_bf16_wmma<<<gBig, 256, 0, stream>>>(mk, Wk_b, kbuf, NTOK_, H_, H_);
  gemm_bf16_wmma<<<gBig, 256, 0, stream>>>(mv, Wv_b, vbuf, NTOK_, H_, H_);

  // decay branch: tanh(xw@w1)@w2  -> wdec -> exp(-exp(...))
  const size_t n128 = (size_t)NTOK_ * 128;
  gemm_bf16_wmma<<<dim3(1, NTOK_ / 128), 256, 0, stream>>>(mw, w1p, t1, NTOK_, 128, H_);
  tanh_bf16_kernel<<<g1d(n128), TB, 0, stream>>>(t1, t1b, n128);
  gemm_bf16_wmma<<<dim3(H_ / 128, NTOK_ / 128), 256, 0, stream>>>(t1b, w2p, wdec, NTOK_, H_, 128);
  decay_kernel<<<g1d(NE_), TB, 0, stream>>>(wdec, w0, NE_);

  // a branch: sigmoid(a0 + (xa@a1)@a2)
  gemm_bf16_wmma<<<dim3(1, NTOK_ / 128), 256, 0, stream>>>(ma, a1p, t1, NTOK_, 128, H_);
  cast_bf16_kernel<<<g1d(n128), TB, 0, stream>>>(t1, t1b, n128);
  gemm_bf16_wmma<<<dim3(H_ / 128, NTOK_ / 128), 256, 0, stream>>>(t1b, a2p, abuf, NTOK_, H_, 128);
  afinal_kernel<<<g1d(NE_), TB, 0, stream>>>(abuf, a0, NE_);

  // v residual mix: v += (vfirst - v) * sigmoid(v0 + (xv@v1)@v2)
  gemm_bf16_wmma<<<dim3(1, NTOK_ / 128), 256, 0, stream>>>(mv, v1p, t1, NTOK_, 128, H_);
  cast_bf16_kernel<<<g1d(n128), TB, 0, stream>>>(t1, t1b, n128);
  gemm_bf16_wmma<<<dim3(H_ / 128, NTOK_ / 128), 256, 0, stream>>>(t1b, v2p, tmpE, NTOK_, H_, 128);
  vupdate_kernel<<<g1d(NE_), TB, 0, stream>>>(vbuf, tmpE, vfirst, v0, NE_);

  // gate branch: g = sigmoid(xg@g1) @ g2
  const size_t n256 = (size_t)NTOK_ * 256;
  gemm_bf16_wmma<<<dim3(256 / 128, NTOK_ / 128), 256, 0, stream>>>(mg, g1p, t1, NTOK_, 256, H_);
  sig_bf16_kernel<<<g1d(n256), TB, 0, stream>>>(t1, t1b, n256);
  gemm_bf16_wmma<<<dim3(H_ / 128, NTOK_ / 128), 256, 0, stream>>>(t1b, g2p, gbuf, NTOK_, H_, 256);

  // kk normalize + k rescale
  kk_kernel<<<NTOK_ * NH_, 64, 0, stream>>>(kbuf, kkbuf, abuf, k_k, k_a);

  // sequential WKV scan; state out goes straight into d_out tuple slot 2
  const size_t oXlast = NE_;
  const size_t oState = oXlast + (size_t)B_ * H_;
  const size_t oVfirst = oState + (size_t)B_ * NH_ * HS_ * HS_;
  wkv_scan_kernel<<<B_ * NH_, 64, 0, stream>>>(rbuf, wdec, kbuf, vbuf, kkbuf, abuf,
                                               wkv_in, ybuf, dout + oState);

  // GroupNorm + rk*v + gate -> bf16
  gn_out_kernel<<<NTOK_ * NH_, 64, 0, stream>>>(ybuf, rbuf, kbuf, vbuf, gbuf,
                                                r_k, ln_w, ln_b, yg);

  // output projection (WMMA) straight into d_out slot 0
  gemm_bf16_wmma<<<gBig, 256, 0, stream>>>(yg, Wo_b, dout, NTOK_, H_, H_);

  // tuple slots 1 and 3
  copy_lastx_kernel<<<g1d((size_t)B_ * H_), TB, 0, stream>>>(x, dout + oXlast);
  copy_f32_kernel<<<g1d(NE_), TB, 0, stream>>>(vfirst, dout + oVfirst, NE_);
}
